// Head_24601572671826
// MI455X (gfx1250) — matrix operand order repaired and checked by reading, not compiled, not run on hardware
//
#include <hip/hip_runtime.h>
#include <stdint.h>

#define B_    8
#define T_    4096
#define C_    1024
#define HS_   64
#define HALF_ 32
#define NEG_INF (-1e30f)

typedef _Float16 h16;
typedef __attribute__((ext_vector_type(16))) _Float16 v16h;
typedef __attribute__((ext_vector_type(8)))  _Float16 v8h;
typedef __attribute__((ext_vector_type(8)))  float    v8f;
typedef __attribute__((ext_vector_type(4)))  float    v4f;
typedef uint32_t su4 __attribute__((ext_vector_type(4)));
typedef uint32_t su8 __attribute__((ext_vector_type(8)));

// ---------------- WMMA fragment helpers (CDNA5 ISA 7.12.2, wave32) ----------------

// A fragment, 16x32 f16, row-major f16 source, leading dim `ld` halfs.
static __device__ inline v16h loadA_rm(const h16* base, int ld, int kc, int lane) {
  const int m  = lane & 15;
  const int kg = (lane >> 4) << 4;
  const h16* p = base + (size_t)m * ld + kc + kg;
  const v8h* p8 = reinterpret_cast<const v8h*>(p);
  v8h lo = p8[0];                      // K = kg+0..7
  v8h hi = p8[1];                      // K = kg+8..15
  v16h a;
#pragma unroll
  for (int i = 0; i < 8; ++i) { a[i] = lo[i]; a[8 + i] = hi[i]; }
  return a;
}

// A fragment from an f32 row-major LDS tile (convert to f16 in-register).
static __device__ inline v16h loadA_ldsf32(const float* xs, int ld, int kc, int lane) {
  const int m  = lane & 15;
  const int kg = (lane >> 4) << 4;
  const float* p = xs + (size_t)m * ld + kc + kg;
  v4f a0 = *reinterpret_cast<const v4f*>(p);
  v4f a1 = *reinterpret_cast<const v4f*>(p + 4);
  v4f a2 = *reinterpret_cast<const v4f*>(p + 8);
  v4f a3 = *reinterpret_cast<const v4f*>(p + 12);
  v16h a;
#pragma unroll
  for (int i = 0; i < 4; ++i) {
    a[i]      = (h16)a0[i];
    a[4 + i]  = (h16)a1[i];
    a[8 + i]  = (h16)a2[i];
    a[12 + i] = (h16)a3[i];
  }
  return a;
}

// B fragment where the lane's 16 K-values are CONTIGUOUS at p (32B -> 2x b128).
static __device__ inline v16h loadB16(const h16* p) {
  const v8h* p8 = reinterpret_cast<const v8h*>(p);
  v8h lo = p8[0], hi = p8[1];
  v16h b;
#pragma unroll
  for (int i = 0; i < 8; ++i) { b[i] = lo[i]; b[8 + i] = hi[i]; }
  return b;
}

static __device__ inline v8f wmma_f16(v16h a, v16h b, v8f c) {
  return __builtin_amdgcn_wmma_f32_16x16x32_f16(false, a, false, b, (short)0, c,
                                                false, false);
}

// ---------------- CDNA5 async / TDM data movement ----------------

static __device__ inline void async_b128_to_lds(uint32_t lds_addr, uint64_t gaddr) {
  asm volatile("global_load_async_to_lds_b128 %0, %1, off"
               :: "v"(lds_addr), "v"(gaddr) : "memory");
}
static __device__ inline void wait_asynccnt0() {
#if __has_builtin(__builtin_amdgcn_s_wait_asynccnt)
  __builtin_amdgcn_s_wait_asynccnt(0);
#else
  asm volatile("s_wait_asynccnt 0" ::: "memory");
#endif
}
static __device__ inline void wait_tensorcnt0() {
#if __has_builtin(__builtin_amdgcn_s_wait_tensorcnt)
  __builtin_amdgcn_s_wait_tensorcnt(0);
#else
  asm volatile("s_wait_tensorcnt 0" ::: "memory");
#endif
}

// TDM: load a rows x elems_x tile of a (tensor_h x tensor_w) f32 tensor into LDS.
// D# layout per CDNA5 ISA ch.8 (group0: 128b, group1: 256b, groups2/3 zero: 2D).
static __device__ __attribute__((always_inline)) void tdm_load_2d_f32(
    uint32_t lds_off, const float* tile_start,
    uint32_t elems_x, uint32_t rows, uint32_t tensor_w, uint32_t tensor_h) {
  const uint64_t ga = (uint64_t)(uintptr_t)tile_start;
  su4 g0;
  g0[0] = 1u;                                            // count=1 (user D#)
  g0[1] = lds_off;                                       // lds_addr (bytes)
  g0[2] = (uint32_t)(ga & 0xffffffffu);                  // global_addr[31:0]
  g0[3] = (uint32_t)((ga >> 32) & 0x01ffffffu)           // global_addr[56:32]
        | 0x80000000u;                                   // type=2 ("image")
  su8 g1;
  g1[0] = 2u << 16;                                      // data_size=2 -> 4 bytes
  g1[1] = (tensor_w & 0xffffu) << 16;                    // tensor_dim0[15:0]
  g1[2] = ((tensor_w >> 16) & 0xffffu)                   // tensor_dim0[31:16]
        | ((tensor_h & 0xffffu) << 16);                  // tensor_dim1[15:0]
  g1[3] = ((tensor_h >> 16) & 0xffffu)                   // tensor_dim1[31:16]
        | ((elems_x & 0xffffu) << 16);                   // tile_dim0
  g1[4] = rows & 0xffffu;                                // tile_dim1 (tile_dim2=0)
  g1[5] = tensor_w;                                      // tensor_dim0_stride[31:0]
  g1[6] = 0;                                             // stride0[47:32], stride1[15:0]
  g1[7] = 0;
  su4 g2 = {0u, 0u, 0u, 0u};
  su4 g3 = {0u, 0u, 0u, 0u};
  asm volatile("tensor_load_to_lds %0, %1, %2, %3"
               :: "s"(g0), "s"(g1), "s"(g2), "s"(g3) : "memory");
}

// ---------------- Kernel 0: pack projection weights into lane-ordered fragments ----
__global__ __launch_bounds__(128) void wpack_kernel(
    const float* __restrict__ Wq, const float* __restrict__ Wk,
    const float* __restrict__ Wv,
    h16* __restrict__ qp, h16* __restrict__ kp, h16* __restrict__ vp) {
  const int w    = blockIdx.x * 4 + (threadIdx.x >> 5);   // 0..383
  const int lane = threadIdx.x & 31;
  const int mat  = w >> 7;                                // 0..2
  const int rem  = w & 127;
  const int n0t  = rem >> 5;                              // 0..3
  const int kct  = rem & 31;                              // 0..31
  const float* W = (mat == 0) ? Wq : (mat == 1) ? Wk : Wv;
  h16*         P = (mat == 0) ? qp : (mat == 1) ? kp : vp;

  const int n  = n0t * 16 + (lane & 15);
  const int kb = kct * 32 + ((lane >> 4) << 4);
  v8h lo, hi;
#pragma unroll
  for (int j = 0; j < 8; ++j) {
    lo[j] = (h16)W[(size_t)(kb + j) * HS_ + n];
    hi[j] = (h16)W[(size_t)(kb + 8 + j) * HS_ + n];
  }
  h16* dst = P + (((size_t)n0t * 32 + kct) * 32 + lane) * 16;
  *reinterpret_cast<v8h*>(dst)     = lo;
  *reinterpret_cast<v8h*>(dst + 8) = hi;
}

// ---------------- Kernel 1: QKV projection + RoPE ----------------
// grid = B*T/16 blocks, 128 threads (4 waves); wave w -> cols [16w,16w+16).
// x tile staged to LDS via the Tensor Data Mover in two 512-col phases;
// RoPE cos/sin rows async-copied to LDS at entry (waited only at the end).
__global__ __launch_bounds__(128) void qkv_rope_kernel(
    const float* __restrict__ x,    const float* __restrict__ cosT,
    const float* __restrict__ sinT,
    const h16* __restrict__ qWp, const h16* __restrict__ kWp,
    const h16* __restrict__ vWp,
    h16* __restrict__ q_ws, h16* __restrict__ k_ws, h16* __restrict__ v_ws) {
  __shared__ float xs[16 * 512];        // 32 KB, TDM destination
  __shared__ float csn[2 * 16 * HALF_]; // cos rows then sin rows (4 KB)
  const int tid  = threadIdx.x;
  const int r0   = blockIdx.x * 16;     // flat row = b*T + t
  const int t0   = r0 & (T_ - 1);
  const int bIdx = r0 >> 12;            // /T_
  const int lane = tid & 31;
  const int n0t  = tid >> 5;            // wave id = n-tile

  // async-stage cos/sin rows [t0, t0+16) (2 KB each); consumed after GEMM loop
  {
    const uint32_t lc = (uint32_t)(uintptr_t)(&csn[0]) + (uint32_t)tid * 16u;
    async_b128_to_lds(lc,
        (uint64_t)(uintptr_t)(cosT + (size_t)t0 * HALF_ + tid * 4));
    async_b128_to_lds(lc + 2048u,
        (uint64_t)(uintptr_t)(sinT + (size_t)t0 * HALF_ + tid * 4));
  }

  v8f cq = {}, ck = {}, cv = {};

  for (int ph = 0; ph < 2; ++ph) {
    if (tid < 32) {   // one TDM issue per workgroup (wave 0); TENSORcnt tracked
      tdm_load_2d_f32((uint32_t)(uintptr_t)xs,
                      x + (size_t)r0 * C_ + ph * 512,
                      /*elems_x=*/512, /*rows=*/16,
                      /*tensor_w=*/C_, /*tensor_h=*/B_ * T_);
      wait_tensorcnt0();
    }
    __syncthreads();

    for (int kc = 0; kc < 512; kc += 32) {
      const int kct = (ph * 512 + kc) >> 5;
      const size_t fo = (((size_t)n0t * 32 + kct) * 32 + lane) * 16;
      const v16h a  = loadA_ldsf32(xs, 512, kc, lane);
      const v16h bq = loadB16(qWp + fo);
      const v16h bk = loadB16(kWp + fo);
      const v16h bv = loadB16(vWp + fo);
      cq = wmma_f16(a, bq, cq);
      ck = wmma_f16(a, bk, ck);
      cv = wmma_f16(a, bv, cv);
    }
    __syncthreads();
  }

  wait_asynccnt0();      // cos/sin staging complete (per-wave) ...
  __syncthreads();       // ... and visible to all waves

  // ---- RoPE in-register; partner column d^1 lives in lane^1 ----
  const int  d    = n0t * 16 + (lane & 15);
  const int  mh   = (lane >> 4) << 3;
  const int  pidx = d >> 1;
  const bool odd  = d & 1;
  v8h vvec;
#pragma unroll
  for (int i = 0; i < 8; ++i) {
    const float cs = csn[(mh + i) * HALF_ + pidx];
    const float sn = csn[512 + (mh + i) * HALF_ + pidx];
    const float oq = __shfl_xor(cq[i], 1, 32);
    const float ok = __shfl_xor(ck[i], 1, 32);
    const float rq = odd ? (oq * sn + cq[i] * cs) : (cq[i] * cs - oq * sn);
    const float rk = odd ? (ok * sn + ck[i] * cs) : (ck[i] * cs - ok * sn);
    const size_t gr = (size_t)(r0 + mh + i);
    q_ws[gr * HS_ + d] = (h16)rq;                       // [b*T+t][d] row-major
    k_ws[gr * HS_ + d] = (h16)rk;                       // [b*T+t][d] row-major
    vvec[i] = (h16)cv[i];
  }
  // V stored d-major [b][d][t]: lane's 8 t-values contiguous -> one b128 store
  *reinterpret_cast<v8h*>(v_ws + ((size_t)bIdx * HS_ + d) * T_ + t0 + mh) = vvec;
}

// ---------------- Kernel 2: causal flash attention ----------------
__global__ __launch_bounds__(32) void flash_attn_kernel(
    const h16* __restrict__ q_ws, const h16* __restrict__ k_ws,
    const h16* __restrict__ v_ws, float* __restrict__ out) {
  __shared__ float Srow[16 * 32];
  __shared__ h16   Pbuf[16 * 32];
  __shared__ float rowM[16], rowL[16], rowF[16];

  const int lane = threadIdx.x;
  const int bIdx = blockIdx.x >> 8;        // T_/16 = 256 tiles per batch
  const int tq0  = (blockIdx.x & 255) * 16;

  const h16* qBase = q_ws + ((size_t)bIdx * T_ + tq0) * HS_;
  const h16* kBase = k_ws + (size_t)bIdx * T_ * HS_;     // row-major [t][d]
  const h16* vBase = v_ws + (size_t)bIdx * HS_ * T_;     // d-major  [d][t]

  const v16h qA0 = loadA_rm(qBase, HS_, 0,  lane);
  const v16h qA1 = loadA_rm(qBase, HS_, 32, lane);

  if (lane < 16) { rowM[lane] = NEG_INF; rowL[lane] = 0.f; }
  __syncthreads();

  v8f o0 = {}, o1 = {}, o2 = {}, o3 = {};
  const int   nkb   = (tq0 + 16 + 31) >> 5;
  const int   mh    = (lane >> 4) << 3;
  const int   nl    = lane & 15;
  const int   khL   = (lane >> 4) << 4;    // 0 / 16 : lane-half K offset
  const float scale = 0.125f;              // 1/sqrt(64)

  for (int jb = 0; jb < nkb; ++jb) {
    const int kb0 = jb << 5;

    // ---- S = Q @ K^T : B[k=d][n=key] = K[key][d], contiguous in d ----
    v8f s0 = {}, s1 = {};
    s0 = wmma_f16(qA0, loadB16(kBase + (size_t)(kb0 + nl) * HS_ + 0  + khL), s0);
    s0 = wmma_f16(qA1, loadB16(kBase + (size_t)(kb0 + nl) * HS_ + 32 + khL), s0);
    s1 = wmma_f16(qA0, loadB16(kBase + (size_t)(kb0 + 16 + nl) * HS_ + 0  + khL), s1);
    s1 = wmma_f16(qA1, loadB16(kBase + (size_t)(kb0 + 16 + nl) * HS_ + 32 + khL), s1);
    if (jb + 1 < nkb) {
      __builtin_prefetch(kBase + (size_t)(kb0 + 32) * HS_, 0, 1);
      __builtin_prefetch(vBase + (size_t)(kb0 + 32), 0, 1);
    }
#pragma unroll
    for (int i = 0; i < 8; ++i) {
      Srow[(mh + i) * 32 + nl]      = s0[i] * scale;
      Srow[(mh + i) * 32 + nl + 16] = s1[i] * scale;
    }
    __syncthreads();

    // ---- online softmax: lane m owns row m ----
    if (lane < 16) {
      const int m  = lane;
      const int tq = tq0 + m;
      const float mOld = rowM[m];
      float mNew = mOld;
      float sv[32];
#pragma unroll
      for (int n = 0; n < 32; ++n) {
        const float s = (kb0 + n <= tq) ? Srow[m * 32 + n] : NEG_INF;
        sv[n] = s;
        mNew = fmaxf(mNew, s);
      }
      const float f = (mNew > 0.5f * NEG_INF) ? __expf(mOld - mNew) : 1.0f;
      float psum = 0.f;
#pragma unroll
      for (int n = 0; n < 32; ++n) {
        const float p = (sv[n] > 0.5f * NEG_INF) ? __expf(sv[n] - mNew) : 0.f;
        psum += p;
        Pbuf[m * 32 + n] = (h16)p;
      }
      rowL[m] = rowL[m] * f + psum;
      rowM[m] = mNew;
      rowF[m] = f;
    }
    __syncthreads();

    // ---- rescale O, then O += P @ V : B[k=key][n=d] = vT[d][key] ----
#pragma unroll
    for (int i = 0; i < 8; ++i) {
      const float f = rowF[mh + i];
      o0[i] *= f; o1[i] *= f; o2[i] *= f; o3[i] *= f;
    }
    const v16h aP = loadA_rm(Pbuf, 32, 0, lane);
    o0 = wmma_f16(aP, loadB16(vBase + (size_t)(0  + nl) * T_ + kb0 + khL), o0);
    o1 = wmma_f16(aP, loadB16(vBase + (size_t)(16 + nl) * T_ + kb0 + khL), o1);
    o2 = wmma_f16(aP, loadB16(vBase + (size_t)(32 + nl) * T_ + kb0 + khL), o2);
    o3 = wmma_f16(aP, loadB16(vBase + (size_t)(48 + nl) * T_ + kb0 + khL), o3);
    __syncthreads();
  }

  // ---- normalize and store f32 output ----
#pragma unroll
  for (int i = 0; i < 8; ++i) {
    const float inv = 1.0f / rowL[mh + i];
    const size_t rowOff = ((size_t)bIdx * T_ + tq0 + mh + i) * HS_;
    out[rowOff + nl]      = o0[i] * inv;
    out[rowOff + 16 + nl] = o1[i] * inv;
    out[rowOff + 32 + nl] = o2[i] * inv;
    out[rowOff + 48 + nl] = o3[i] * inv;
  }
}

extern "C" void kernel_launch(void* const* d_in, const int* in_sizes, int n_in,
                              void* d_out, int out_size, void* d_ws, size_t ws_size,
                              hipStream_t stream) {
  (void)in_sizes; (void)n_in; (void)out_size; (void)ws_size;
  const float* x    = (const float*)d_in[0];
  const float* cosT = (const float*)d_in[1];
  const float* sinT = (const float*)d_in[2];
  const float* Wq   = (const float*)d_in[3];
  const float* Wk   = (const float*)d_in[4];
  const float* Wv   = (const float*)d_in[5];
  float* out = (float*)d_out;

  const size_t nQKV  = (size_t)B_ * T_ * HS_;          // elems per q/k/v
  const size_t nPack = (size_t)4 * 32 * 32 * 16;       // elems per packed W
  char* ws = (char*)d_ws;
  h16* q_ws = (h16*)ws;
  h16* k_ws = (h16*)(ws + nQKV * 2);
  h16* v_ws = (h16*)(ws + nQKV * 4);
  h16* qWp  = (h16*)(ws + nQKV * 6);
  h16* kWp  = (h16*)(ws + nQKV * 6 + nPack * 2);
  h16* vWp  = (h16*)(ws + nQKV * 6 + nPack * 4);

  wpack_kernel<<<dim3(96), dim3(128), 0, stream>>>(Wq, Wk, Wv, qWp, kWp, vWp);
  qkv_rope_kernel<<<dim3(B_ * T_ / 16), dim3(128), 0, stream>>>(
      x, cosT, sinT, qWp, kWp, vWp, q_ws, k_ws, v_ws);
  flash_attn_kernel<<<dim3(B_ * (T_ / 16)), dim3(32), 0, stream>>>(
      q_ws, k_ws, v_ws, out);
}
